// DAWNBlock_12979391168722
// MI455X (gfx1250) — matrix operand order repaired and checked by reading, not compiled, not run on hardware
//
#include <hip/hip_runtime.h>
#include <hip/hip_bf16.h>

// ---------------------------------------------------------------------------
// DAWN neuron-routing block for MI455X (gfx1250, wave32, WMMA).
//
// Sizes: B=8, S=4096, D_MODEL=1024, D_SPACE=64.
// Only neurons [0, 3840) are consumed by the reference (qk 2048 + v 1024 +
// rel 512 + val 256); the 4096 "know" neurons are dead code -> skipped.
//
// Roofline: proj GEMM reads 128MB of x (memory bound, fp32 WMMA 16x16x4);
// logits GEMM is 2x16 GFLOP compute (bf16 WMMA 16x16x32, fp32 accum).
// ---------------------------------------------------------------------------

typedef __attribute__((ext_vector_type(2)))  float  v2f;
typedef __attribute__((ext_vector_type(8)))  float  v8f;
typedef __attribute__((ext_vector_type(16))) __bf16 v16bf;

#define D_SPACE   64
#define N_USED    3840   // qk+v+rel+val
#define BATCH     8
#define SEQ       4096
#define ROWS      (BATCH * SEQ)   // 32768

__device__ __forceinline__ unsigned short f2bf(float f) {
    unsigned int u = __float_as_uint(f);
    unsigned int r = u + 0x7FFFu + ((u >> 16) & 1u);   // round-to-nearest-even
    return (unsigned short)(r >> 16);
}

union BF16x16 { v16bf v; uint4 q[2]; };

// ---------------------------------------------------------------------------
// prep: normalize neuron_emb rows (first 3840) and store bf16, row-major n x 64
// ---------------------------------------------------------------------------
__global__ void prep_emb_kernel(const float* __restrict__ emb,
                                unsigned short* __restrict__ emb_bf) {
    int n = blockIdx.x * blockDim.x + threadIdx.x;
    if (n >= N_USED) return;
    const float* e = emb + (size_t)n * D_SPACE;
    float ss = 0.f;
    #pragma unroll 8
    for (int k = 0; k < D_SPACE; ++k) { float v = e[k]; ss += v * v; }
    float inv = rsqrtf(ss);
    unsigned short* o = emb_bf + (size_t)n * D_SPACE;
    #pragma unroll 8
    for (int k = 0; k < D_SPACE; ++k) o[k] = f2bf(e[k] * inv);
}

__global__ void zero_w_kernel(float* __restrict__ w, int n) {
    int i = blockIdx.x * blockDim.x + threadIdx.x;
    if (i < n) w[i] = 0.f;
}

// ---------------------------------------------------------------------------
// proj: h[row][0:64] = x[row][:] @ W_proj + b_proj, written as bf16.
// One wave = one 16-row M-tile; 4 N-tiles of 16; K=1024 stepped by 4 with
// V_WMMA_F32_16X16X4_F32 (fp32 in, fp32 accum).
//   A 16x4 f32 frag: lane(0-15)=rows, VGPRj = K(2*kh+j), kh=lane>>4
//   B 4x16  f32 frag: lane%16=col,     VGPRj = K(2*kh+j)
// ---------------------------------------------------------------------------
__global__ __launch_bounds__(256) void proj_kernel(
        const float* __restrict__ x, const float* __restrict__ W,
        const float* __restrict__ bias, unsigned short* __restrict__ h_bf) {
    const int tid  = threadIdx.x;
    const int wave = tid >> 5;
    const int lane = tid & 31;
    const int kh   = lane >> 4;
    const int ln   = lane & 15;
    const int rowBase = (blockIdx.x * 8 + wave) * 16;
    const float* xp = x + (size_t)(rowBase + ln) * 1024;

    v8f c0 = {}, c1 = {}, c2 = {}, c3 = {};
    for (int k = 0; k < 1024; k += 4) {
        v2f a;
        a.x = xp[k + 2 * kh];
        a.y = xp[k + 2 * kh + 1];
        const float* wp = W + (size_t)(k + 2 * kh) * 64 + ln;
        v2f b0, b1, b2, b3;
        b0.x = wp[0];  b0.y = wp[64];
        b1.x = wp[16]; b1.y = wp[80];
        b2.x = wp[32]; b2.y = wp[96];
        b3.x = wp[48]; b3.y = wp[112];
        c0 = __builtin_amdgcn_wmma_f32_16x16x4_f32(false, a, false, b0, (short)0, c0, false, false);
        c1 = __builtin_amdgcn_wmma_f32_16x16x4_f32(false, a, false, b1, (short)0, c1, false, false);
        c2 = __builtin_amdgcn_wmma_f32_16x16x4_f32(false, a, false, b2, (short)0, c2, false, false);
        c3 = __builtin_amdgcn_wmma_f32_16x16x4_f32(false, a, false, b3, (short)0, c3, false, false);
    }
    // C layout: lanes0-15 -> M=r, lanes16-31 -> M=r+8; col = nt*16 + ln
    #pragma unroll
    for (int nt = 0; nt < 4; ++nt) {
        v8f c = (nt == 0) ? c0 : (nt == 1) ? c1 : (nt == 2) ? c2 : c3;
        int col = nt * 16 + ln;
        float bv = bias[col];
        #pragma unroll
        for (int r = 0; r < 8; ++r) {
            int row = rowBase + kh * 8 + r;
            h_bf[(size_t)row * 64 + col] = f2bf(c[r] + bv);
        }
    }
}

// ---------------------------------------------------------------------------
// logits + segment softmax + importance-weighted accumulation.
// One wave = 16 rows (A frags resident: K=64 -> two 16x16x32 bf16 frags).
// Per segment: pass1 accumulates per-row sum(exp(logit)) (logits bounded, so
// max-free softmax is exact); pass2 recomputes the WMMAs and deposits
// sum_s importance[s] * prob[s,n] into LDS, then one global atomic per col.
// ---------------------------------------------------------------------------
__device__ __forceinline__ void load_emb_frags(const unsigned short* __restrict__ emb_bf,
                                               int ncol, int kh, BF16x16& b0, BF16x16& b1) {
    const unsigned short* ep = emb_bf + (size_t)ncol * 64;
    b0.q[0] = *(const uint4*)(ep + kh * 8);
    b0.q[1] = *(const uint4*)(ep + 16 + kh * 8);
    b1.q[0] = *(const uint4*)(ep + 32 + kh * 8);
    b1.q[1] = *(const uint4*)(ep + 48 + kh * 8);
}

__global__ __launch_bounds__(256) void logits_kernel(
        const unsigned short* __restrict__ h_bf,
        const unsigned short* __restrict__ emb_bf,
        const float* __restrict__ importance,
        float* __restrict__ wacc) {
    __shared__ float wloc[N_USED];
    const int tid = threadIdx.x;
    for (int i = tid; i < N_USED; i += 256) wloc[i] = 0.f;
    __syncthreads();

    const int wave = tid >> 5;
    const int lane = tid & 31;
    const int kh   = lane >> 4;
    const int ln   = lane & 15;
    const int rowBase = blockIdx.x * 128 + wave * 16;   // 128 rows/WG, same batch
    const int b = (blockIdx.x * 128) / SEQ;

    // A frags (16x32 bf16 layout): lane row = ln; K-halves select by kh.
    const unsigned short* hp = h_bf + (size_t)(rowBase + ln) * 64;
    BF16x16 a0, a1;
    a0.q[0] = *(const uint4*)(hp + kh * 8);
    a0.q[1] = *(const uint4*)(hp + 16 + kh * 8);
    a1.q[0] = *(const uint4*)(hp + 32 + kh * 8);
    a1.q[1] = *(const uint4*)(hp + 48 + kh * 8);

    // This lane's C rows are rowBase + kh*8 + r (r = VGPR slot 0..7)
    float imp8[8];
    const float* ip = importance + rowBase + kh * 8;
    #pragma unroll
    for (int r = 0; r < 8; ++r) imp8[r] = ip[r];

    const int segBase[4]  = {0, 2048, 3072, 3584};
    const int segTiles[4] = {128, 64, 32, 16};

    #pragma unroll 1
    for (int s = 0; s < 4; ++s) {
        const int base   = segBase[s];
        const int ntiles = segTiles[s];

        // ---- pass 1: per-row exp-sums (per-lane partials over own columns)
        float psum[8];
        #pragma unroll
        for (int r = 0; r < 8; ++r) psum[r] = 0.f;
        for (int t = 0; t < ntiles; ++t) {
            BF16x16 b0, b1;
            load_emb_frags(emb_bf, base + t * 16 + ln, kh, b0, b1);
            v8f c = {};
            c = __builtin_amdgcn_wmma_f32_16x16x32_bf16(false, a0.v, false, b0.v, (short)0, c, false, false);
            c = __builtin_amdgcn_wmma_f32_16x16x32_bf16(false, a1.v, false, b1.v, (short)0, c, false, false);
            #pragma unroll
            for (int r = 0; r < 8; ++r) psum[r] += __expf(c[r]);
        }
        // butterfly-reduce within each 16-lane half (rows differ across halves)
        float isum[8];
        #pragma unroll
        for (int r = 0; r < 8; ++r) {
            float ssum = psum[r];
            ssum += __shfl_xor(ssum, 1, 32);
            ssum += __shfl_xor(ssum, 2, 32);
            ssum += __shfl_xor(ssum, 4, 32);
            ssum += __shfl_xor(ssum, 8, 32);
            isum[r] = 1.0f / ssum;
        }

        // ---- pass 2: recompute logits, deposit importance-weighted probs
        for (int t = 0; t < ntiles; ++t) {
            BF16x16 b0, b1;
            load_emb_frags(emb_bf, base + t * 16 + ln, kh, b0, b1);
            v8f c = {};
            c = __builtin_amdgcn_wmma_f32_16x16x32_bf16(false, a0.v, false, b0.v, (short)0, c, false, false);
            c = __builtin_amdgcn_wmma_f32_16x16x32_bf16(false, a1.v, false, b1.v, (short)0, c, false, false);
            float contrib = 0.f;
            #pragma unroll
            for (int r = 0; r < 8; ++r) contrib += __expf(c[r]) * isum[r] * imp8[r];
            contrib += __shfl_xor(contrib, 16, 32);  // fold the two row-halves
            if (lane < 16) atomicAdd(&wloc[base + t * 16 + lane], contrib);
        }
    }
    __syncthreads();
    for (int i = tid; i < N_USED; i += 256)
        atomicAdd(&wacc[(size_t)b * N_USED + i], wloc[i]);
}

// ---------------------------------------------------------------------------
// finalize: per-batch top-k selection + output assembly.
// Out layout (floats): idx_qk[8*64] | idx_v[8*32] | rel_Q[8*512] |
//                      rel_K[8*512] | val_w[8*256]   (total 11008)
// ---------------------------------------------------------------------------
__global__ __launch_bounds__(256) void finalize_kernel(
        const float* __restrict__ wacc, float* __restrict__ out) {
    __shared__ float sv[2048];
    __shared__ float rbv[256];
    __shared__ int   rbi[256];
    __shared__ float pickv[64];
    __shared__ int   picki[64];
    const int tid = threadIdx.x;
    const int b   = blockIdx.x;

    const int segBase[4] = {0, 2048, 3072, 3584};
    const int segCnt[4]  = {2048, 1024, 512, 256};
    const int segK[4]    = {64, 32, 16, 3};

    for (int s = 0; s < 4; ++s) {
        const int base = segBase[s], cnt = segCnt[s], K = segK[s];
        for (int i = tid; i < cnt; i += 256) sv[i] = wacc[(size_t)b * N_USED + base + i];
        __syncthreads();
        for (int k = 0; k < K; ++k) {
            float bm = -1e30f; int bi = 0x7fffffff;
            for (int i = tid; i < cnt; i += 256) {
                float v = sv[i];
                if (v > bm || (v == bm && i < bi)) { bm = v; bi = i; }
            }
            rbv[tid] = bm; rbi[tid] = bi;
            __syncthreads();
            if (tid == 0) {
                float best = -1e30f; int besti = 0x7fffffff;
                for (int t = 0; t < 256; ++t)
                    if (rbv[t] > best || (rbv[t] == best && rbi[t] < besti)) { best = rbv[t]; besti = rbi[t]; }
                picki[k] = besti; pickv[k] = best;
                sv[besti] = -1e30f;
            }
            __syncthreads();
        }
        if (s == 0 || s == 1) {
            if (tid == 0) {  // sort indices ascending (jnp.sort in reference)
                for (int i = 1; i < K; ++i) {
                    int key = picki[i], j = i - 1;
                    while (j >= 0 && picki[j] > key) { picki[j + 1] = picki[j]; --j; }
                    picki[j + 1] = key;
                }
            }
            __syncthreads();
            int off = (s == 0 ? 0 : 512) + b * K;
            for (int k = tid; k < K; k += 256) out[off + k] = (float)picki[k];
        } else if (s == 2) {
            for (int i = tid; i < 512; i += 256) {
                out[768 + b * 512 + i]  = 0.f;   // rel_Q
                out[4864 + b * 512 + i] = 0.f;   // rel_K (identical values)
            }
            __syncthreads();
            if (tid < K) {
                int idx = picki[tid]; float v = pickv[tid];
                out[768 + b * 512 + idx]  = v;
                out[4864 + b * 512 + idx] = v;
            }
        } else {
            for (int i = tid; i < 256; i += 256) out[8960 + b * 256 + i] = 0.f;  // val_w
            __syncthreads();
            if (tid < K) out[8960 + b * 256 + picki[tid]] = pickv[tid];
        }
        __syncthreads();
    }
}

// ---------------------------------------------------------------------------
extern "C" void kernel_launch(void* const* d_in, const int* in_sizes, int n_in,
                              void* d_out, int out_size, void* d_ws, size_t ws_size,
                              hipStream_t stream) {
    (void)in_sizes; (void)n_in; (void)out_size; (void)ws_size;
    const float* x          = (const float*)d_in[0];
    const float* importance = (const float*)d_in[1];
    const float* W_proj     = (const float*)d_in[2];
    const float* b_proj     = (const float*)d_in[3];
    const float* neuron_emb = (const float*)d_in[4];
    float* out = (float*)d_out;

    char* ws = (char*)d_ws;
    unsigned short* h_bf   = (unsigned short*)ws;                              // 4 MiB: 32768 x 64 bf16
    unsigned short* emb_bf = (unsigned short*)(ws + (4u << 20));               // 480 KiB: 3840 x 64 bf16
    float*          wacc   = (float*)(ws + (4u << 20) + (512u << 10));         // 120 KiB: 8 x 3840 f32

    prep_emb_kernel<<<(N_USED + 255) / 256, 256, 0, stream>>>(neuron_emb, emb_bf);
    zero_w_kernel<<<(BATCH * N_USED + 255) / 256, 256, 0, stream>>>(wacc, BATCH * N_USED);
    proj_kernel<<<ROWS / 128, 256, 0, stream>>>(x, W_proj, b_proj, h_bf);      // 256 blocks
    logits_kernel<<<ROWS / 128, 256, 0, stream>>>(h_bf, emb_bf, importance, wacc);
    finalize_kernel<<<BATCH, 256, 0, stream>>>(wacc, out);
}
